// MultiheadAttention_82978768159155
// MI455X (gfx1250) — matrix operand order repaired and checked by reading, not compiled, not run on hardware
//
#include <hip/hip_runtime.h>
#include <hip/hip_bf16.h>

// ---------------------------------------------------------------------------
// LoRA multi-head attention forward for MI455X (gfx1250, wave32, WMMA bf16).
// Double-buffered LDS + async global->LDS copies (ASYNCcnt), 64x64 wave tiles.
// ---------------------------------------------------------------------------

typedef __bf16 bhalf;
typedef __attribute__((ext_vector_type(16))) __bf16        v16bf;
typedef __attribute__((ext_vector_type(8)))  float         v8f;
typedef __attribute__((ext_vector_type(4)))  unsigned int  u32x4;
typedef __attribute__((ext_vector_type(4)))  int           v4i;

#define D_MODEL   2048
#define NUM_HEADS 16
#define D_K       128
#define SEQ       2048
#define BATCH     2
#define M_TOTAL   (BATCH * SEQ)      // 4096 rows
#define LORA_R    8
#define LORA_SCALING 2.0f
#define AUG_K     32                 // padded lora K-tail (8 real + 24 zero)

union FragBF { u32x4 u[2]; v16bf v; };   // 32 bytes = 16 bf16

// --------- CDNA5 async global->LDS path (guarded; host pass falls back) -----
#if defined(__HIP_DEVICE_COMPILE__) &&                                   \
    __has_builtin(__builtin_amdgcn_global_load_async_to_lds_b128) &&     \
    __has_builtin(__builtin_amdgcn_s_wait_asynccnt)
#define ASYNC_COPY 1
#define AS_GLOBAL __attribute__((address_space(1)))
#define AS_LDS    __attribute__((address_space(3)))
#define ASYNC_CP_B128(gsrc, ldst)                                        \
    __builtin_amdgcn_global_load_async_to_lds_b128(                      \
        (AS_GLOBAL v4i*)(gsrc), (AS_LDS v4i*)(ldst), 0, 0)
#define WAIT_ASYNC() __builtin_amdgcn_s_wait_asynccnt(0)
#else
#define ASYNC_COPY 0
#define ASYNC_CP_B128(gsrc, ldst)                                        \
    (*(u32x4*)(ldst) = *(const u32x4*)(gsrc))
#define WAIT_ASYNC() ((void)0)
#endif

// ------------------------------ helpers ------------------------------------

__global__ void cvt_f32_to_bf16(const float* __restrict__ in,
                                bhalf* __restrict__ out, int n) {
    int i = blockIdx.x * blockDim.x + threadIdx.x;
    int stride = gridDim.x * blockDim.x;
    for (; i < n; i += stride) out[i] = (bhalf)in[i];
}

// Pack lora B [D_MODEL][8] fp32 -> [D_MODEL][32] bf16 (zero padded)
__global__ void pack_lora_B(const float* __restrict__ Bm,
                            bhalf* __restrict__ out) {
    int t = blockIdx.x * blockDim.x + threadIdx.x;
    if (t >= D_MODEL * AUG_K) return;
    int row = t >> 5, c = t & 31;
    out[t] = (c < LORA_R) ? (bhalf)Bm[row * LORA_R + c] : (bhalf)0.0f;
}

// augX[m][r] = SCALING * dot(src[m,:], A[r,:])   (r<8; cols 8..31 zero)
__global__ void lora_xA(const bhalf* __restrict__ src,
                        const float* __restrict__ A,
                        bhalf* __restrict__ out, int M) {
    int t = blockIdx.x * blockDim.x + threadIdx.x;
    if (t >= M * AUG_K) return;
    int m = t >> 5, c = t & 31;
    if (c >= LORA_R) { out[t] = (bhalf)0.0f; return; }
    const bhalf* xr = src + (size_t)m * D_MODEL;
    const float* ar = A + (size_t)c * D_MODEL;
    float acc = 0.f;
    for (int k = 0; k < D_MODEL; k += 4) {
        acc += (float)xr[k + 0] * ar[k + 0] + (float)xr[k + 1] * ar[k + 1]
             + (float)xr[k + 2] * ar[k + 2] + (float)xr[k + 3] * ar[k + 3];
    }
    out[t] = (bhalf)(acc * LORA_SCALING);
}

// --------------------------- WMMA GEMM (+LoRA tail) -------------------------
// out[m][n] = sum_k Abuf[m][k]*Wbuf[n][k] + bias[n] + tail(augX,augB)
// Block tile 128x128, 4 waves (2x2), wave tile 64x64 -> 4x4 wmma accumulators.
// Per K step: 16 ds_load_b128 feed 16 WMMAs (1.0 wmma per LDS load).
// Software pipeline: async-stage tile kt+1 into ping-pong LDS while computing
// tile kt; one s_wait_asynccnt + one barrier per K step.

#define BT_M 128
#define BT_N 128
#define KT   32
#define LDSS 40      // padded LDS row stride (bf16 elements)

template <bool OUT_BF16>
__global__ void __launch_bounds__(128)
gemm_lora(const bhalf* __restrict__ Abuf,   // [M][2048] bf16
          const bhalf* __restrict__ Wbuf,   // [2048][2048] bf16 (row n = W[n,:])
          const float* __restrict__ bias,   // [2048]
          const bhalf* __restrict__ Aaug,   // [M][32]
          const bhalf* __restrict__ Baug,   // [2048][32]
          float* __restrict__ outF,
          bhalf* __restrict__ outB) {
    __shared__ __align__(16) bhalf As[2][BT_M * LDSS];
    __shared__ __align__(16) bhalf Bs[2][BT_N * LDSS];

    const int tid  = threadIdx.x;
    const int lane = tid & 31;
    const int wave = tid >> 5;         // 0..3
    const int wm   = wave >> 1;        // 0..1 (M direction)
    const int wn   = wave & 1;         // 0..1 (N direction)
    const int l15  = lane & 15;
    const int half = lane >> 4;

    const int rowBase = blockIdx.y * BT_M;
    const int colBase = blockIdx.x * BT_N;

    v8f zero = {0.f, 0.f, 0.f, 0.f, 0.f, 0.f, 0.f, 0.f};
    v8f acc[4][4];
    for (int i = 0; i < 4; ++i)
        for (int j = 0; j < 4; ++j) acc[i][j] = zero;

    const int nKT = D_MODEL / KT;      // 64 dense K tiles + 1 lora tail tile

    auto srcPtrs = [&](int kt, const bhalf*& aSrc, const bhalf*& bSrc,
                       int& stride) {
        if (kt < nKT) {
            aSrc = Abuf + (size_t)rowBase * D_MODEL + kt * KT;
            bSrc = Wbuf + (size_t)colBase * D_MODEL + kt * KT;
            stride = D_MODEL;
        } else {
            aSrc = Aaug + (size_t)rowBase * AUG_K;
            bSrc = Baug + (size_t)colBase * AUG_K;
            stride = AUG_K;
        }
    };

    // Stage one 128x32 tile pair: 512 chunks of 16B each, 128 threads x 4.
    auto stageTile = [&](int buf, const bhalf* aSrc, const bhalf* bSrc,
                         int stride) {
        for (int i = 0; i < 4; ++i) {
            int e  = tid + i * 128;
            int r  = e >> 2;
            int c8 = (e & 3) * 8;
            ASYNC_CP_B128(aSrc + (size_t)r * stride + c8,
                          &As[buf][r * LDSS + c8]);
            ASYNC_CP_B128(bSrc + (size_t)r * stride + c8,
                          &Bs[buf][r * LDSS + c8]);
        }
    };

    // Prologue: stage tile 0.
    {
        const bhalf *a0, *b0; int st0;
        srcPtrs(0, a0, b0, st0);
        stageTile(0, a0, b0, st0);
    }
    WAIT_ASYNC();
    __syncthreads();

    for (int kt = 0; kt <= nKT; ++kt) {
        const int cur = kt & 1;

        // Kick off async stage of the next tile into the other buffer.
        if (kt < nKT) {
            const bhalf *an, *bn; int stn;
            srcPtrs(kt + 1, an, bn, stn);
            stageTile(cur ^ 1, an, bn, stn);
        }

        // Batch all 8 fragment loads (16 ds_load_b128), then 16 WMMAs.
        FragBF bfr[4], afr[4];
        for (int ni = 0; ni < 4; ++ni) {
            const bhalf* p =
                &Bs[cur][(wn * 64 + ni * 16 + l15) * LDSS + half * 8];
            bfr[ni].u[0] = *(const u32x4*)p;
            bfr[ni].u[1] = *(const u32x4*)(p + 16);
        }
        for (int mi = 0; mi < 4; ++mi) {
            const bhalf* p = &As[cur][(wm * 64 + mi * 16 + l15) * LDSS];
            afr[mi].u[0] = *(const u32x4*)(p + half * 8);
            afr[mi].u[1] = *(const u32x4*)(p + 16 + half * 8);
        }
        for (int mi = 0; mi < 4; ++mi)
            for (int ni = 0; ni < 4; ++ni)
                acc[mi][ni] = __builtin_amdgcn_wmma_f32_16x16x32_bf16(
                    false, afr[mi].v, false, bfr[ni].v, (short)0,
                    acc[mi][ni], false, false);

        WAIT_ASYNC();       // next tile landed in LDS (this wave's copies)
        __syncthreads();    // all waves: reads of cur done, next tile visible
    }

    // Epilogue: bias add + store. C layout: row = j + half*8, col = lane&15.
    for (int mi = 0; mi < 4; ++mi) {
        for (int ni = 0; ni < 4; ++ni) {
            int col = colBase + wn * 64 + ni * 16 + l15;
            float bv = bias[col];
            for (int j = 0; j < 8; ++j) {
                int row = rowBase + wm * 64 + mi * 16 + half * 8 + j;
                float v = acc[mi][ni][j] + bv;
                if (OUT_BF16)
                    outB[(size_t)row * D_MODEL + col] = (bhalf)v;
                else
                    outF[(size_t)row * D_MODEL + col] = v;
            }
        }
    }
}

// ----------------------------- Flash attention ------------------------------
// Grid: (BATCH*NUM_HEADS, SEQ/64). Block: 128 threads = 4 waves, each wave
// owns 16 query rows. Online softmax, bf16 WMMA for both QK^T and P*V.

#define ATT_BR 64
#define ATT_BC 64
#define QK_SCALE 0.08838834764831845f   // 1/sqrt(128)
#define QLD 136   // Q/K LDS row stride (bf16)
#define VLD 72    // Vt / P LDS row stride (bf16)

__global__ void __launch_bounds__(128)
flash_attn(const bhalf* __restrict__ Qg, const bhalf* __restrict__ Kg,
           const bhalf* __restrict__ Vg, bhalf* __restrict__ Ctx) {
    __shared__ __align__(16) bhalf Qs[ATT_BR * QLD];
    __shared__ __align__(16) bhalf Ks[ATT_BC * QLD];
    __shared__ __align__(16) bhalf Vt[D_K * VLD];        // transposed V tile
    __shared__ __align__(16) bhalf Ps[4][16 * VLD];      // per-wave P tile

    const int tid  = threadIdx.x;
    const int lane = tid & 31;
    const int wave = tid >> 5;
    const int l15  = lane & 15;
    const int half = lane >> 4;

    const int bh = blockIdx.x;           // 0..31
    const int b  = bh >> 4;
    const int h  = bh & 15;
    const int q0 = blockIdx.y * ATT_BR;

    const size_t rowStart = (size_t)b * SEQ;  // row offset into [4096][2048]
    const int colH = h * D_K;

    // Async-stage Q tile [64][128] once (1024 chunks of 8 bf16 / 128 thr).
    for (int i = 0; i < 8; ++i) {
        int e  = tid + i * 128;
        int r  = e >> 4;
        int c8 = (e & 15) * 8;
        ASYNC_CP_B128(Qg + (rowStart + q0 + r) * D_MODEL + colH + c8,
                      &Qs[r * QLD + c8]);
    }

    v8f zero = {0.f, 0.f, 0.f, 0.f, 0.f, 0.f, 0.f, 0.f};
    float mrun[8], lrun[8];
    v8f o[8];
    for (int j = 0; j < 8; ++j) { mrun[j] = -1e30f; lrun[j] = 0.f; }
    for (int nt = 0; nt < 8; ++nt) o[nt] = zero;

    for (int kt = 0; kt < SEQ / ATT_BC; ++kt) {
        const int k0 = kt * ATT_BC;
        __syncthreads();   // previous tile's LDS reads complete
        // K tile [64][128] via async copy; V tile transposed manually.
        for (int i = 0; i < 8; ++i) {
            int e  = tid + i * 128;
            int r  = e >> 4;
            int c8 = (e & 15) * 8;
            ASYNC_CP_B128(Kg + (rowStart + k0 + r) * D_MODEL + colH + c8,
                          &Ks[r * QLD + c8]);
            u32x4 tmp =
                *(const u32x4*)(Vg + (rowStart + k0 + r) * D_MODEL + colH + c8);
            const bhalf* tv = (const bhalf*)&tmp;
            for (int t2 = 0; t2 < 8; ++t2) Vt[(c8 + t2) * VLD + r] = tv[t2];
        }
        WAIT_ASYNC();      // covers Q (first iteration) + this K tile
        __syncthreads();

        // --- S = Q * K^T for this wave's 16 rows (16x64, fp32 accum) ---
        v8f s[4];
        for (int ni = 0; ni < 4; ++ni) s[ni] = zero;
        for (int kc = 0; kc < 4; ++kc) {            // dk in chunks of 32
            // Batch 1 A-fragment + 4 B-fragments, then 4 WMMAs.
            FragBF af, bfr[4];
            const bhalf* ap = &Qs[(wave * 16 + l15) * QLD + kc * 32];
            af.u[0] = *(const u32x4*)(ap + half * 8);
            af.u[1] = *(const u32x4*)(ap + 16 + half * 8);
            for (int ni = 0; ni < 4; ++ni) {
                const bhalf* bp =
                    &Ks[(ni * 16 + l15) * QLD + kc * 32 + half * 8];
                bfr[ni].u[0] = *(const u32x4*)bp;
                bfr[ni].u[1] = *(const u32x4*)(bp + 16);
            }
            for (int ni = 0; ni < 4; ++ni)
                s[ni] = __builtin_amdgcn_wmma_f32_16x16x32_bf16(
                    false, af.v, false, bfr[ni].v, (short)0, s[ni],
                    false, false);
        }

        // --- online softmax (row-wise, lanes 0..15 / 16..31 hold N=0..15) ---
        float rescale[8];
        for (int j = 0; j < 8; ++j) {
            float mx = s[0][j] * QK_SCALE;
            for (int ni = 1; ni < 4; ++ni)
                mx = fmaxf(mx, s[ni][j] * QK_SCALE);
            for (int msk = 1; msk < 16; msk <<= 1)
                mx = fmaxf(mx, __shfl_xor(mx, msk, 32));
            float mnew = fmaxf(mrun[j], mx);
            rescale[j] = __expf(mrun[j] - mnew);
            mrun[j] = mnew;
            float psum = 0.f;
            for (int ni = 0; ni < 4; ++ni) {
                float p = __expf(s[ni][j] * QK_SCALE - mnew);
                s[ni][j] = p;
                psum += p;
            }
            for (int msk = 1; msk < 16; msk <<= 1)
                psum += __shfl_xor(psum, msk, 32);
            lrun[j] = lrun[j] * rescale[j] + psum;
            for (int nt = 0; nt < 8; ++nt) o[nt][j] *= rescale[j];
        }

        // --- P -> per-wave LDS (bf16) to re-layout as A fragments ---
        bhalf* pw = &Ps[wave][0];
        for (int ni = 0; ni < 4; ++ni)
            for (int j = 0; j < 8; ++j)
                pw[(half * 8 + j) * VLD + ni * 16 + l15] = (bhalf)s[ni][j];

        // --- O += P * V  (K = 64 keys -> 2 chunks of 32) ---
        for (int kc = 0; kc < 2; ++kc) {
            FragBF af;
            const bhalf* ap = &pw[l15 * VLD + kc * 32];
            af.u[0] = *(const u32x4*)(ap + half * 8);
            af.u[1] = *(const u32x4*)(ap + 16 + half * 8);
            // Two batches of 4 B-fragments to bound VGPR pressure.
            for (int g = 0; g < 2; ++g) {
                FragBF bfr[4];
                for (int q = 0; q < 4; ++q) {
                    int nt = g * 4 + q;
                    const bhalf* bp =
                        &Vt[(nt * 16 + l15) * VLD + kc * 32 + half * 8];
                    bfr[q].u[0] = *(const u32x4*)bp;
                    bfr[q].u[1] = *(const u32x4*)(bp + 16);
                }
                for (int q = 0; q < 4; ++q) {
                    int nt = g * 4 + q;
                    o[nt] = __builtin_amdgcn_wmma_f32_16x16x32_bf16(
                        false, af.v, false, bfr[q].v, (short)0, o[nt],
                        false, false);
                }
            }
        }
    }

    // Normalize and write context tile (bf16 for the O-projection GEMM).
    for (int j = 0; j < 8; ++j) {
        float inv = 1.0f / lrun[j];
        int row = q0 + wave * 16 + half * 8 + j;
        for (int nt = 0; nt < 8; ++nt) {
            int col = colH + nt * 16 + l15;
            Ctx[(rowStart + row) * D_MODEL + col] = (bhalf)(o[nt][j] * inv);
        }
    }
}

// ------------------------------- launcher -----------------------------------

extern "C" void kernel_launch(void* const* d_in, const int* in_sizes, int n_in,
                              void* d_out, int out_size, void* d_ws,
                              size_t ws_size, hipStream_t stream) {
    (void)in_sizes; (void)n_in; (void)out_size; (void)ws_size;

    const float* x = (const float*)d_in[0];
    const float* W[4] = { (const float*)d_in[1], (const float*)d_in[5],
                          (const float*)d_in[9], (const float*)d_in[13] };
    const float* bias[4] = { (const float*)d_in[2], (const float*)d_in[6],
                             (const float*)d_in[10], (const float*)d_in[14] };
    const float* Al[4] = { (const float*)d_in[3], (const float*)d_in[7],
                           (const float*)d_in[11], (const float*)d_in[15] };
    const float* Bl[4] = { (const float*)d_in[4], (const float*)d_in[8],
                           (const float*)d_in[12], (const float*)d_in[16] };
    float* out = (float*)d_out;

    // ---- workspace carve-up ----
    char* ws = (char*)d_ws;
    size_t off = 0;
    auto carve = [&](size_t bytes) -> char* {
        char* p = ws + off;
        off += (bytes + 255) & ~(size_t)255;
        return p;
    };
    const size_t xBytes = (size_t)M_TOTAL * D_MODEL * sizeof(bhalf);
    const size_t wBytes = (size_t)D_MODEL * D_MODEL * sizeof(bhalf);

    bhalf* xb = (bhalf*)carve(xBytes);
    bhalf* Wb[4];
    for (int i = 0; i < 4; ++i) Wb[i] = (bhalf*)carve(wBytes);
    bhalf* qb  = (bhalf*)carve(xBytes);
    bhalf* kb  = (bhalf*)carve(xBytes);
    bhalf* vb  = (bhalf*)carve(xBytes);
    bhalf* ctx = (bhalf*)carve(xBytes);
    bhalf* augX[4];
    for (int i = 0; i < 4; ++i)
        augX[i] = (bhalf*)carve((size_t)M_TOTAL * AUG_K * sizeof(bhalf));
    bhalf* augB[4];
    for (int i = 0; i < 4; ++i)
        augB[i] = (bhalf*)carve((size_t)D_MODEL * AUG_K * sizeof(bhalf));

    // ---- 1. fp32 -> bf16 conversions ----
    {
        int n = M_TOTAL * D_MODEL;
        cvt_f32_to_bf16<<<(n + 255) / 256, 256, 0, stream>>>(x, xb, n);
        int nw = D_MODEL * D_MODEL;
        for (int i = 0; i < 4; ++i)
            cvt_f32_to_bf16<<<(nw + 255) / 256, 256, 0, stream>>>(W[i], Wb[i],
                                                                  nw);
        for (int i = 0; i < 4; ++i)
            pack_lora_B<<<(D_MODEL * AUG_K + 255) / 256, 256, 0, stream>>>(
                Bl[i], augB[i]);
    }

    // ---- 2. LoRA xA tails for q,k,v ----
    {
        int n = M_TOTAL * AUG_K;
        for (int i = 0; i < 3; ++i)
            lora_xA<<<(n + 255) / 256, 256, 0, stream>>>(xb, Al[i], augX[i],
                                                         M_TOTAL);
    }

    // ---- 3. QKV projections (bf16 WMMA GEMM, bf16 outputs) ----
    dim3 gGrid(D_MODEL / BT_N, M_TOTAL / BT_M);   // (16, 32)
    gemm_lora<true><<<gGrid, 128, 0, stream>>>(xb, Wb[0], bias[0], augX[0],
                                               augB[0], nullptr, qb);
    gemm_lora<true><<<gGrid, 128, 0, stream>>>(xb, Wb[1], bias[1], augX[1],
                                               augB[1], nullptr, kb);
    gemm_lora<true><<<gGrid, 128, 0, stream>>>(xb, Wb[2], bias[2], augX[2],
                                               augB[2], nullptr, vb);

    // ---- 4. flash attention -> context (bf16) ----
    dim3 aGrid(BATCH * NUM_HEADS, SEQ / ATT_BR);  // (32, 32)
    flash_attn<<<aGrid, 128, 0, stream>>>(qb, kb, vb, ctx);

    // ---- 5. LoRA tail for O projection (from context) ----
    {
        int n = M_TOTAL * AUG_K;
        lora_xA<<<(n + 255) / 256, 256, 0, stream>>>(ctx, Al[3], augX[3],
                                                     M_TOTAL);
    }

    // ---- 6. output projection (fp32 output into d_out) ----
    gemm_lora<false><<<gGrid, 128, 0, stream>>>(ctx, Wb[3], bias[3], augX[3],
                                                augB[3], out, nullptr);
}
